// QuantizedLinear_63342177681752
// MI455X (gfx1250) — compile-verified, matrix operand order checked by reading, not run on hardware
//
#include <hip/hip_runtime.h>

typedef __attribute__((ext_vector_type(16))) _Float16 v16h;
typedef __attribute__((ext_vector_type(8)))  _Float16 v8h;
typedef __attribute__((ext_vector_type(4)))  _Float16 v4h;
typedef __attribute__((ext_vector_type(8)))  float    v8f;
typedef __attribute__((ext_vector_type(4)))  float    v4f;
typedef __attribute__((ext_vector_type(4)))  int      v4i;

#define TILE_M 128
#define TILE_N 128
#define TILE_K 32
#define LDS_STRIDE 48   // halves: 96B row pitch -> 16B-aligned 8-half chunks, no bank conflicts
#define M_TOTAL 4096    // B*S = 2*2048
#define N_TOTAL 4096    // D_OUT
#define K_TOTAL 4096    // D_IN
#define NSTEPS  (K_TOTAL / TILE_K)

__global__ __launch_bounds__(256)
void qlinear_wmma_f16split(const float* __restrict__ x,
                           const int*   __restrict__ qw,
                           const float* __restrict__ scale,
                           const int*   __restrict__ zp,
                           const float* __restrict__ bias,
                           float*       __restrict__ out)
{
    // ping-pong double buffers: 2 * 3 * 128*48*2B = 72 KB (of 320 KB/WGP)
    __shared__ __align__(16) _Float16 sAh[2][TILE_M * LDS_STRIDE];
    __shared__ __align__(16) _Float16 sAl[2][TILE_M * LDS_STRIDE];
    __shared__ __align__(16) _Float16 sB [2][TILE_N * LDS_STRIDE];

    const int tid  = threadIdx.x;
    const int lane = tid & 31;
    const int wave = tid >> 5;      // 0..7 (wave32)
    const int wm   = wave >> 1;     // 0..3 : 32-row slab
    const int wn   = wave & 1;      // 0..1 : 64-col slab

    const int mbase = blockIdx.y * TILE_M;
    const int nbase = blockIdx.x * TILE_N;

    // cooperative loader indexing: 128 rows x 8 groups of 4 = 1024 vec4s, 4/thread
    const int lrow0 = tid >> 3;         // 0..31
    const int lcol  = (tid & 7) * 4;    // 0,4,...,28

    // per-thread zero points for the 4 weight rows this thread streams (constant over K)
    int zpt[4];
#pragma unroll
    for (int i = 0; i < 4; ++i)
        zpt[i] = zp[nbase + lrow0 + i * 32];

    // base pointers for this thread's streaming slots
    const float* xbase[4];
    const int*   wbase[4];
#pragma unroll
    for (int i = 0; i < 4; ++i) {
        xbase[i] = x  + (size_t)(mbase + lrow0 + i * 32) * K_TOTAL + lcol;
        wbase[i] = qw + (size_t)(nbase + lrow0 + i * 32) * K_TOTAL + lcol;
    }

    v8f acc[2][4] = {};

    // staging registers for the in-flight global tile
    v4f xs[4];
    v4i ws[4];

    // ---- prologue: fetch tile 0 and park it in buffer 0 ----
#pragma unroll
    for (int i = 0; i < 4; ++i) {
        xs[i] = *(const v4f*)(xbase[i]);
        ws[i] = *(const v4i*)(wbase[i]);
    }
#pragma unroll
    for (int i = 0; i < 4; ++i) {
        const int row = lrow0 + i * 32;
        v4h hh, ll, wv;
#pragma unroll
        for (int e = 0; e < 4; ++e) {
            _Float16 h = (_Float16)xs[i][e];
            hh[e] = h;
            ll[e] = (_Float16)(xs[i][e] - (float)h);
            wv[e] = (_Float16)(float)(ws[i][e] - zpt[i]);
        }
        *(v4h*)&sAh[0][row * LDS_STRIDE + lcol] = hh;
        *(v4h*)&sAl[0][row * LDS_STRIDE + lcol] = ll;
        *(v4h*)&sB [0][row * LDS_STRIDE + lcol] = wv;
    }
    __syncthreads();

    // fragment gather pattern (documented 16-bit A/B layout):
    // lanes 0-15 : row M=lane,    K chunks {0..7, 16..23}
    // lanes 16-31: row M=lane-16, K chunks {8..15, 24..31}
    const int frow = lane & 15;
    const int fk   = (lane >> 4) * 8;   // 0 or 8

    for (int t = 0; t < NSTEPS; ++t) {
        const int cur = t & 1;
        const bool have_next = (t + 1) < NSTEPS;

        // ---- issue next tile's global loads first (overlap with WMMA) ----
        if (have_next) {
            const int koff = (t + 1) * TILE_K;
#pragma unroll
            for (int i = 0; i < 4; ++i) {
                xs[i] = *(const v4f*)(xbase[i] + koff);
                ws[i] = *(const v4i*)(wbase[i] + koff);
            }
            // L2 hint two tiles ahead
            if (t + 2 < NSTEPS) {
                __builtin_prefetch(xbase[0] + (t + 2) * TILE_K, 0, 0);
                __builtin_prefetch(wbase[0] + (t + 2) * TILE_K, 0, 0);
            }
        }

        // ---- fragment gather from current buffer ----
        v16h ah[2], al[2], bf[4];
#pragma unroll
        for (int i = 0; i < 2; ++i) {
            const _Float16* p = &sAh[cur][(wm * 32 + i * 16 + frow) * LDS_STRIDE + fk];
            ((v8h*)&ah[i])[0] = *(const v8h*)p;
            ((v8h*)&ah[i])[1] = *(const v8h*)(p + 16);
            const _Float16* q = &sAl[cur][(wm * 32 + i * 16 + frow) * LDS_STRIDE + fk];
            ((v8h*)&al[i])[0] = *(const v8h*)q;
            ((v8h*)&al[i])[1] = *(const v8h*)(q + 16);
        }
#pragma unroll
        for (int j = 0; j < 4; ++j) {
            const _Float16* p = &sB[cur][(wn * 64 + j * 16 + frow) * LDS_STRIDE + fk];
            ((v8h*)&bf[j])[0] = *(const v8h*)p;
            ((v8h*)&bf[j])[1] = *(const v8h*)(p + 16);
        }

        // ---- 16 WMMAs / wave / K-step; B fragment reused across hi/lo passes ----
#pragma unroll
        for (int i = 0; i < 2; ++i)
#pragma unroll
            for (int j = 0; j < 4; ++j) {
                acc[i][j] = __builtin_amdgcn_wmma_f32_16x16x32_f16(
                    false, ah[i], false, bf[j], (short)0, acc[i][j], false, false);
                acc[i][j] = __builtin_amdgcn_wmma_f32_16x16x32_f16(
                    false, al[i], false, bf[j], (short)0, acc[i][j], false, false);
            }

        // ---- convert + park next tile in the other buffer; one barrier/iter ----
        // (writes target buf cur^1, whose last readers were ordered by the
        //  previous iteration's barrier -> no extra barrier needed before stores)
        if (have_next) {
            const int nxt = cur ^ 1;
#pragma unroll
            for (int i = 0; i < 4; ++i) {
                const int row = lrow0 + i * 32;
                v4h hh, ll, wv;
#pragma unroll
                for (int e = 0; e < 4; ++e) {
                    _Float16 h = (_Float16)xs[i][e];
                    hh[e] = h;
                    ll[e] = (_Float16)(xs[i][e] - (float)h);
                    wv[e] = (_Float16)(float)(ws[i][e] - zpt[i]);
                }
                *(v4h*)&sAh[nxt][row * LDS_STRIDE + lcol] = hh;
                *(v4h*)&sAl[nxt][row * LDS_STRIDE + lcol] = ll;
                *(v4h*)&sB [nxt][row * LDS_STRIDE + lcol] = wv;
            }
            __syncthreads();
        }
    }

    // ---- epilogue: y = acc * scale[n] + bias[n] ----
    // C/D layout: VGPR r, lanes 0-15 -> (M=r, N=lane); lanes 16-31 -> (M=r+8, N=lane-16)
    const int mhalf = (lane < 16) ? 0 : 8;
#pragma unroll
    for (int j = 0; j < 4; ++j) {
        const int n = nbase + wn * 64 + j * 16 + frow;
        const float s = scale[n];
        const float b = bias[n];
#pragma unroll
        for (int i = 0; i < 2; ++i) {
            const int m0 = mbase + wm * 32 + i * 16 + mhalf;
#pragma unroll
            for (int r = 0; r < 8; ++r)
                out[(size_t)(m0 + r) * N_TOTAL + n] = acc[i][j][r] * s + b;
        }
    }
}

extern "C" void kernel_launch(void* const* d_in, const int* in_sizes, int n_in,
                              void* d_out, int out_size, void* d_ws, size_t ws_size,
                              hipStream_t stream) {
    (void)in_sizes; (void)n_in; (void)out_size; (void)d_ws; (void)ws_size;
    const float* x     = (const float*)d_in[0];
    const int*   qw    = (const int*)  d_in[1];
    const float* scale = (const float*)d_in[2];
    const int*   zp    = (const int*)  d_in[3];
    const float* bias  = (const float*)d_in[4];
    float*       out   = (float*)d_out;

    dim3 grid(N_TOTAL / TILE_N, M_TOTAL / TILE_M);  // 32 x 32 blocks
    dim3 block(256);                                // 8 wave32 waves
    qlinear_wmma_f16split<<<grid, block, 0, stream>>>(x, qw, scale, zp, bias, out);
}
